// OscillatorLayer_22393959481783
// MI455X (gfx1250) — matrix operand order repaired and checked by reading, not compiled, not run on hardware
//
#include <hip/hip_runtime.h>

typedef __attribute__((ext_vector_type(16))) _Float16 v16h;
typedef __attribute__((ext_vector_type(8)))  float    v8f;

#define NE   8388608   // B*C*H*W
#define Bn   8
#define Cn   64
#define Hn   128
#define Wn   128
#define HWn  16384
#define ODT  0.3f
#define LRELU(x) ((x) > 0.f ? (x) : 0.2f*(x))

// workspace half-region fragment offsets (in halfs)
#define PW_F   0        // pw_w       : 4 Mtiles * 2 Kfrags = 8 frags
#define H1_F   4096     // h_w1[:, :64]: 2*2 = 4 frags
#define H2_F   6144     // h_w2       : 2*1 = 2 frags
#define H3_F   7168     // h_w3       : 16*1 = 16 frags
#define RP1_F  15360    // r_pw1_w    : 8 frags
#define RP2_F  19456    // r_pw2_w    : 8 frags
// workspace byte offsets
#define B1S_OFF   49152
#define MU_OFF    (49152 + 128)
#define RSG_OFF   (49152 + 128 + 2048)
#define MAPS_OFF  65536

// ---- WMMA fragment helpers (C/D layout per CDNA5 ISA 7.12.2; A/B use contiguous k order) ----

// A matrix 16x32 f16: lane L holds row M = L%16; elems 0..15 -> K = kb..kb+15,
// kb = (L<16 ? 0 : 16).  Pre-swizzled in ws.
__device__ __forceinline__ v16h load_A(const _Float16* frags, int fragIdx, int lane) {
  return *(const v16h*)(frags + ((size_t)fragIdx * 32 + lane) * 16);
}

// B matrix 32x16 f16 from LDS tile T[pix][stride]: lane L holds col N = L%16,
// K = (L<16 ? 0..15 : 16..31) + kf*32 (contiguous 16 halfs -> aligned load).
__device__ __forceinline__ v16h load_B(const _Float16* T, int stride, int nt, int kf, int lane) {
  int pix = nt * 16 + (lane & 15);
  int kb  = kf * 32 + ((lane & 16) ? 16 : 0);
  return *(const v16h*)(T + pix * stride + kb);
}

// C/D 16x16 f32: VGPR r, lane L -> M = r + (L<16?0:8), N = L%16.
__device__ __forceinline__ v8f bias_C(const float* bias, int mt, int lane) {
  v8f c;
  int mb = mt * 16 + ((lane & 16) ? 8 : 0);
#pragma unroll
  for (int r = 0; r < 8; ++r) c[r] = bias[mb + r];
  return c;
}

#define WMMA(a, b, c) __builtin_amdgcn_wmma_f32_16x16x32_f16(false, (a), false, (b), (short)0, (c), false, false)

// ---- prep: swizzle weights into A-fragment layout, fold style into bias ----

__device__ void fill_frags(const float* W, int RS, int Mtiles, int KF, _Float16* dst) {
  int total = Mtiles * KF * 32;
  for (int u = threadIdx.x; u < total; u += blockDim.x) {
    int frag = u >> 5, lane = u & 31;
    int mt = frag / KF, kf = frag - mt * KF;
    int row = mt * 16 + (lane & 15);
    int kb  = kf * 32 + ((lane & 16) ? 16 : 0);
    _Float16* d = dst + ((size_t)frag * 32 + lane) * 16;
#pragma unroll
    for (int j = 0; j < 8; ++j) d[j]     = (_Float16)W[row * RS + kb + j];
#pragma unroll
    for (int j = 0; j < 8; ++j) d[8 + j] = (_Float16)W[row * RS + kb + 8 + j];
  }
}

__global__ void osc_prep_k(const float* h_w1, const float* h_b1, const float* h_w2,
                           const float* h_w3, const float* pw_w, const float* r_pw1_w,
                           const float* r_pw2_w, const float* style,
                           _Float16* wsh, float* b1s) {
  fill_frags(pw_w,    64, 4, 2, wsh + PW_F);
  fill_frags(h_w1,   128, 2, 2, wsh + H1_F);   // force part (cols 0..63)
  fill_frags(h_w2,    32, 2, 1, wsh + H2_F);
  fill_frags(h_w3,    32, 16, 1, wsh + H3_F);
  fill_frags(r_pw1_w, 64, 4, 2, wsh + RP1_F);
  fill_frags(r_pw2_w, 64, 4, 2, wsh + RP2_F);
  if (threadIdx.x < 32) {
    int o = threadIdx.x;
    float s = h_b1[o];
    for (int j = 0; j < 64; ++j) s += h_w1[o * 128 + 64 + j] * style[j];
    b1s[o] = s;
  }
}

// ---- hyper: f = force+0.2*illum; per-pixel MLP -> packed a12 / gb maps ----

__global__ void osc_hyper_k(const float* __restrict__ force, const float* __restrict__ illum,
                            const float* h_b2, const float* h_b3,
                            const _Float16* wsh, const float* b1s,
                            _Float16* a12, _Float16* gb, float* fmap) {
  __shared__ _Float16 xT[32 * 80];
  __shared__ _Float16 u1T[32 * 48];
  __shared__ _Float16 u2T[32 * 48];
  __shared__ float    pbuf[256 * 33];
  const int tid = threadIdx.x, lane = tid & 31, wv = tid >> 5;
  const int wbase = blockIdx.x * 32, h = blockIdx.y, b = blockIdx.z;
  const size_t base0 = (size_t)b * Cn * HWn + (size_t)h * Wn + wbase;

  for (int u = tid; u < 2048; u += 256) {
    int c = u >> 5, pix = u & 31;
    size_t g = base0 + (size_t)c * HWn + pix;
    float val = force[g] + 0.2f * illum[g];
    fmap[g] = val;
    xT[pix * 80 + c] = (_Float16)val;
  }
  __syncthreads();

  if (wv < 4) {  // L1: [32,64] @ [64,32pix]
    int mt = wv >> 1, nt = wv & 1;
    v8f c = bias_C(b1s, mt, lane);
#pragma unroll
    for (int kf = 0; kf < 2; ++kf)
      c = WMMA(load_A(wsh + H1_F, mt * 2 + kf, lane), load_B(xT, 80, nt, kf, lane), c);
#pragma unroll
    for (int r = 0; r < 8; ++r) {
      int ch = mt * 16 + r + ((lane & 16) ? 8 : 0);
      int px = nt * 16 + (lane & 15);
      u1T[px * 48 + ch] = (_Float16)LRELU(c[r]);
    }
  }
  __syncthreads();

  if (wv < 4) {  // L2: [32,32]
    int mt = wv >> 1, nt = wv & 1;
    v8f c = bias_C(h_b2, mt, lane);
    c = WMMA(load_A(wsh + H2_F, mt, lane), load_B(u1T, 48, nt, 0, lane), c);
#pragma unroll
    for (int r = 0; r < 8; ++r) {
      int ch = mt * 16 + r + ((lane & 16) ? 8 : 0);
      int px = nt * 16 + (lane & 15);
      u2T[px * 48 + ch] = (_Float16)LRELU(c[r]);
    }
  }
  __syncthreads();

  for (int j = wv; j < 32; j += 8) {  // L3: [256,32]
    int mt = j >> 1, nt = j & 1;
    v8f c = bias_C(h_b3, mt, lane);
    c = WMMA(load_A(wsh + H3_F, mt, lane), load_B(u2T, 48, nt, 0, lane), c);
#pragma unroll
    for (int r = 0; r < 8; ++r)
      pbuf[(mt * 16 + r + ((lane & 16) ? 8 : 0)) * 33 + nt * 16 + (lane & 15)] = c[r];
  }
  __syncthreads();

  for (int u = tid; u < 2048; u += 256) {
    int c = u >> 5, pix = u & 31;
    float om = 2.f / (1.f + __expf(-pbuf[c * 33 + pix]));
    float ze = 1.f / (1.f + __expf(-pbuf[(64 + c) * 33 + pix]));
    float gg = 2.f / (1.f + __expf(-pbuf[(128 + c) * 33 + pix]));
    float bb = tanhf(pbuf[(192 + c) * 33 + pix]);
    size_t g = base0 + (size_t)c * HWn + pix;
    a12[2 * g]     = (_Float16)(om * om);        // omega^2
    a12[2 * g + 1] = (_Float16)(2.f * ze * om);  // 2*zeta*omega
    gb[2 * g]      = (_Float16)gg;
    gb[2 * g + 1]  = (_Float16)bb;
  }
}

// ---- step 0 specialization: x0 = v0 = 0  ->  v1 = f*DT, x1 = f*DT^2 ----

__global__ void osc_step0_k(const float* __restrict__ fmap,
                            float* __restrict__ xdst, float* __restrict__ vbuf) {
  size_t i = (size_t)blockIdx.x * blockDim.x + threadIdx.x;
  size_t stride = (size_t)gridDim.x * blockDim.x;
  for (; i < (size_t)NE; i += stride) {
    float v = fmap[i] * ODT;
    vbuf[i] = v;
    xdst[i] = v * ODT;
  }
}

// ---- one symplectic-Euler step: coupling = dw3x3(x) + pw_w @ x ----

__global__ void osc_step_k(const float* __restrict__ xsrc, float* __restrict__ xdst,
                           float* __restrict__ vbuf, const float* __restrict__ fmap,
                           const _Float16* __restrict__ a12,
                           const float* dw_k, const _Float16* wsh) {
  __shared__ float    xr[3 * 64 * 36];
  __shared__ float    dwk[576];
  __shared__ _Float16 xT[32 * 80];
  __shared__ float    pwres[64 * 33];
  const int tid = threadIdx.x, lane = tid & 31, wv = tid >> 5;
  const int wbase = blockIdx.x * 32, h = blockIdx.y, b = blockIdx.z;
  const size_t base0 = (size_t)b * Cn * HWn + (size_t)h * Wn + wbase;

  // prefetch update-phase operands (one 128B row per channel) while halo loads
  if (tid < 64) {
    size_t g = base0 + (size_t)tid * HWn;
    __builtin_prefetch(fmap + g, 0, 3);
    __builtin_prefetch(vbuf + g, 1, 3);
    __builtin_prefetch(a12 + 2 * g, 0, 3);
  }
  for (int u = tid; u < 576; u += 256) dwk[u] = dw_k[u];
  for (int u = tid; u < 3 * 64 * 34; u += 256) {
    int r = u / (64 * 34); int rem = u - r * (64 * 34);
    int c = rem / 34;      int cc = rem - c * 34;
    int hh = h - 1 + r;    int w = wbase - 1 + cc;
    float v = 0.f;
    if ((unsigned)hh < (unsigned)Hn && (unsigned)w < (unsigned)Wn)
      v = xsrc[(size_t)b * Cn * HWn + (size_t)c * HWn + (size_t)hh * Wn + w];
    xr[(r * 64 + c) * 36 + cc] = v;
    if (r == 1 && cc >= 1 && cc <= 32)        // fused WMMA B-tile fill (center row)
      xT[(cc - 1) * 80 + c] = (_Float16)v;
  }
  __syncthreads();

  {  // pointwise: pw_w[64,64] @ x[64, 32pix]
    int mt = wv >> 1, nt = wv & 1;
    v8f acc = {};
#pragma unroll
    for (int kf = 0; kf < 2; ++kf)
      acc = WMMA(load_A(wsh + PW_F, mt * 2 + kf, lane), load_B(xT, 80, nt, kf, lane), acc);
#pragma unroll
    for (int r = 0; r < 8; ++r)
      pwres[(mt * 16 + r + ((lane & 16) ? 8 : 0)) * 33 + nt * 16 + (lane & 15)] = acc[r];
  }
  __syncthreads();

  for (int u = tid; u < 2048; u += 256) {
    int c = u >> 5, pix = u & 31;
    float dw = 0.f;
#pragma unroll
    for (int r = 0; r < 3; ++r)
#pragma unroll
      for (int dx = 0; dx < 3; ++dx)
        dw += xr[(r * 64 + c) * 36 + pix + dx] * dwk[c * 9 + r * 3 + dx];
    size_t g = base0 + (size_t)c * HWn + pix;
    const _Float16* pc = a12 + 2 * g;
    float A1 = (float)pc[0], A2 = (float)pc[1];
    float xc = xr[(64 + c) * 36 + pix + 1];
    float v  = vbuf[g];
    float accel = fmap[g] + dw + pwres[c * 33 + pix] - A2 * v - A1 * xc;
    v += accel * ODT;
    vbuf[g] = v;
    xdst[g] = xc + v * ODT;
  }
}

// ---- instance-norm statistics over H,W per (b,c) ----

__global__ void osc_norm_k(const float* __restrict__ fmap, const float* __restrict__ x,
                           float* mu, float* rsg) {
  __shared__ float s1[256], s2[256];
  const int tid = threadIdx.x, bc = blockIdx.x;
  const size_t base = (size_t)bc * HWn;
  float a = 0.f, q = 0.f;
  for (int i = tid; i < HWn; i += 256) {
    float v = fmap[base + i] + x[base + i];
    a += v; q += v * v;
  }
  s1[tid] = a; s2[tid] = q;
  __syncthreads();
  for (int s = 128; s > 0; s >>= 1) {
    if (tid < s) { s1[tid] += s1[tid + s]; s2[tid] += s2[tid + s]; }
    __syncthreads();
  }
  if (tid == 0) {
    float m = s1[0] * (1.f / HWn);
    float var = s2[0] * (1.f / HWn) - m * m;
    mu[bc] = m;
    rsg[bc] = rsqrtf(var + 1e-5f);
  }
}

// ---- final: norm+modulate, refine branch (dw + 2 WMMA pointwise), residual ----

__global__ void osc_final_k(const float* __restrict__ fmap, const float* __restrict__ x,
                            const _Float16* __restrict__ gb,
                            const float* __restrict__ mu, const float* __restrict__ rsg,
                            const float* r_dw_k, const float* r_dw_b,
                            const float* r_pw1_b, const float* r_pw2_b,
                            const float* lscale, const _Float16* wsh,
                            float* __restrict__ out) {
  __shared__ float    nrm[3 * 64 * 36];
  __shared__ float    rdwk[576];
  __shared__ _Float16 rT[32 * 80];
  __shared__ _Float16 r1T[32 * 80];
  const int tid = threadIdx.x, lane = tid & 31, wv = tid >> 5;
  const int wbase = blockIdx.x * 32, h = blockIdx.y, b = blockIdx.z;
  const size_t base0 = (size_t)b * Cn * HWn + (size_t)h * Wn + wbase;

  if (tid < 64) {
    size_t g = base0 + (size_t)tid * HWn;
    __builtin_prefetch(fmap + g, 0, 3);
    __builtin_prefetch(x + g, 0, 3);
    __builtin_prefetch(gb + 2 * g, 0, 3);
  }
  for (int u = tid; u < 576; u += 256) rdwk[u] = r_dw_k[u];
  for (int u = tid; u < 3 * 64 * 34; u += 256) {
    int r = u / (64 * 34); int rem = u - r * (64 * 34);
    int c = rem / 34;      int cc = rem - c * 34;
    int hh = h - 1 + r;    int w = wbase - 1 + cc;
    float v = 0.f;
    if ((unsigned)hh < (unsigned)Hn && (unsigned)w < (unsigned)Wn) {
      size_t g = (size_t)b * Cn * HWn + (size_t)c * HWn + (size_t)hh * Wn + w;
      int bc = b * 64 + c;
      float fx = fmap[g] + x[g];
      const _Float16* pg = gb + 2 * g;
      v = (float)pg[0] * ((fx - mu[bc]) * rsg[bc]) + (float)pg[1];
    }
    nrm[(r * 64 + c) * 36 + cc] = v;
  }
  __syncthreads();

  for (int u = tid; u < 2048; u += 256) {
    int c = u >> 5, pix = u & 31;
    float rv = r_dw_b[c];
#pragma unroll
    for (int r = 0; r < 3; ++r)
#pragma unroll
      for (int dx = 0; dx < 3; ++dx)
        rv += nrm[(r * 64 + c) * 36 + pix + dx] * rdwk[c * 9 + r * 3 + dx];
    rT[pix * 80 + c] = (_Float16)rv;
  }
  __syncthreads();

  {  // r1 = lrelu(r_pw1 @ r + b1)
    int mt = wv >> 1, nt = wv & 1;
    v8f c = bias_C(r_pw1_b, mt, lane);
#pragma unroll
    for (int kf = 0; kf < 2; ++kf)
      c = WMMA(load_A(wsh + RP1_F, mt * 2 + kf, lane), load_B(rT, 80, nt, kf, lane), c);
#pragma unroll
    for (int r = 0; r < 8; ++r) {
      int ch = mt * 16 + r + ((lane & 16) ? 8 : 0);
      int px = nt * 16 + (lane & 15);
      r1T[px * 80 + ch] = (_Float16)LRELU(c[r]);
    }
  }
  __syncthreads();

  {  // r2 = r_pw2 @ r1 + b2 ; out = nrm + layer_scale * r2
    int mt = wv >> 1, nt = wv & 1;
    v8f c = bias_C(r_pw2_b, mt, lane);
#pragma unroll
    for (int kf = 0; kf < 2; ++kf)
      c = WMMA(load_A(wsh + RP2_F, mt * 2 + kf, lane), load_B(r1T, 80, nt, kf, lane), c);
#pragma unroll
    for (int r = 0; r < 8; ++r) {
      int ch = mt * 16 + r + ((lane & 16) ? 8 : 0);
      int px = nt * 16 + (lane & 15);
      size_t g = base0 + (size_t)ch * HWn + px;
      out[g] = nrm[(64 + ch) * 36 + px + 1] + lscale[ch] * c[r];
    }
  }
}

// ---- launcher ----

extern "C" void kernel_launch(void* const* d_in, const int* in_sizes, int n_in,
                              void* d_out, int out_size, void* d_ws, size_t ws_size,
                              hipStream_t stream) {
  (void)in_sizes; (void)n_in; (void)out_size; (void)ws_size;
  const float* force   = (const float*)d_in[0];
  const float* style   = (const float*)d_in[1];
  const float* illum   = (const float*)d_in[2];
  const float* h_w1    = (const float*)d_in[3];
  const float* h_b1    = (const float*)d_in[4];
  const float* h_w2    = (const float*)d_in[5];
  const float* h_b2    = (const float*)d_in[6];
  const float* h_w3    = (const float*)d_in[7];
  const float* h_b3    = (const float*)d_in[8];
  const float* dw_k    = (const float*)d_in[9];
  const float* pw_w    = (const float*)d_in[10];
  const float* r_dw_k  = (const float*)d_in[11];
  const float* r_dw_b  = (const float*)d_in[12];
  const float* r_pw1_w = (const float*)d_in[13];
  const float* r_pw1_b = (const float*)d_in[14];
  const float* r_pw2_w = (const float*)d_in[15];
  const float* r_pw2_b = (const float*)d_in[16];
  const float* lscale  = (const float*)d_in[17];

  float* out = (float*)d_out;
  float* xA  = out + (size_t)NE;       // final x slot (ping)
  float* vB  = out + (size_t)2 * NE;   // v (updated in place)
  float* xS  = out;                    // pong scratch, overwritten by out at the end

  char* ws  = (char*)d_ws;
  _Float16* wsh = (_Float16*)ws;
  float* b1s = (float*)(ws + B1S_OFF);
  float* muv = (float*)(ws + MU_OFF);
  float* rsv = (float*)(ws + RSG_OFF);
  _Float16* a12 = (_Float16*)(ws + MAPS_OFF);            // interleaved {omega^2, 2*zeta*omega}
  _Float16* gb  = a12 + (size_t)2 * NE;                  // interleaved {gamma, beta}
  float* fmap = (float*)(ws + MAPS_OFF + (size_t)8 * NE);

  dim3 grid(Wn / 32, Hn, Bn);
  osc_prep_k<<<1, 256, 0, stream>>>(h_w1, h_b1, h_w2, h_w3, pw_w, r_pw1_w, r_pw2_w,
                                    style, wsh, b1s);
  osc_hyper_k<<<grid, 256, 0, stream>>>(force, illum, h_b2, h_b3, wsh, b1s,
                                        a12, gb, fmap);
  // step 0 closed-form (x0 = v0 = 0): writes x1 -> xS, v1 -> vB
  osc_step0_k<<<4096, 256, 0, stream>>>(fmap, xS, vB);
  // steps 1..9: ping-pong so x10 ends in xA (= d_out[N:2N])
  for (int s = 1; s < 10; ++s) {
    const float* xs = (s & 1) ? xS : xA;
    float*       xd = (s & 1) ? xA : xS;
    osc_step_k<<<grid, 256, 0, stream>>>(xs, xd, vB, fmap, a12, dw_k, wsh);
  }
  osc_norm_k<<<512, 256, 0, stream>>>(fmap, xA, muv, rsv);
  osc_final_k<<<grid, 256, 0, stream>>>(fmap, xA, gb, muv, rsv,
                                        r_dw_k, r_dw_b, r_pw1_b, r_pw2_b, lscale,
                                        wsh, out);
}
